// QKVAttentionLegacy_16518444221163
// MI455X (gfx1250) — matrix operand order repaired and checked by reading, not compiled, not run on hardware
//
#include <hip/hip_runtime.h>

// ---------------------------------------------------------------------------
// Fused multi-head attention for qkv [4, 3072, 1024] fp32, 16 heads, ch=64.
//   per head: q,k,v are [64, 1024] (channel-major, stride 1024)
//   S = (q*s)^T (k*s),  P = softmax(S),  out[c,t] = sum_s P[t,s] * v[c,s]
// Strategy: stage fp32->f16 tiles in LDS, v_wmma_f32_16x16x32_f16 for both
// GEMMs, online (flash) softmax in f32 registers with wave32 shuffles.
// ---------------------------------------------------------------------------

typedef _Float16 f16;
typedef __attribute__((ext_vector_type(8)))  _Float16 v8h;
typedef __attribute__((ext_vector_type(16))) _Float16 v16h;
typedef __attribute__((ext_vector_type(8)))  float    v8f;

#define T_SEQ   1024
#define CH      64
#define BQ      128      // query rows per workgroup (8 waves x 16)
#define BKV     64       // keys per iteration (4 WMMA N-tiles)
#define LDP     72       // padded LDS row stride in halves (144B, 16B aligned)
#define NWAVE   8

__device__ __forceinline__ v8f wmma_f16(v16h a, v16h b, v8f c) {
  return __builtin_amdgcn_wmma_f32_16x16x32_f16(
      /*neg_a=*/false, a, /*neg_b=*/false, b,
      /*c_mod=*/(short)0, c, /*reuse_a=*/false, /*reuse_b=*/false);
}

// A-fragment (16x32 f16, M=row striped per lane): lane half h16 selects the
// K-subrange contiguously: halves 0-15 = K[kb + 16*h16 ..] (matches B).
__device__ __forceinline__ v16h load_a_frag(const f16* row, int kb, int h16) {
  v8h lo = *(const v8h*)(row + kb + h16 * 16);
  v8h hi = *(const v8h*)(row + kb + 8 + h16 * 16);
  return __builtin_shufflevector(lo, hi, 0,1,2,3,4,5,6,7,8,9,10,11,12,13,14,15);
}

// B-fragment (32x16 f16, N=col per lane): lane holds 16 consecutive K values
// for its column; one contiguous 32B read.
__device__ __forceinline__ v16h load_b_frag(const f16* p) {
  v8h lo = *(const v8h*)(p);
  v8h hi = *(const v8h*)(p + 8);
  return __builtin_shufflevector(lo, hi, 0,1,2,3,4,5,6,7,8,9,10,11,12,13,14,15);
}

__global__ __launch_bounds__(256, 2)
void qkv_attn_kernel(const float* __restrict__ qkv, float* __restrict__ out) {
  // LDS staging (f16): Q [128 rows][64 ch], Kt [64 keys][64 ch] (transposed),
  // Vt [64 ch][64 keys], P scratch per wave [16 rows][64 keys].
  __shared__ __align__(16) f16 Qs[BQ * LDP];
  __shared__ __align__(16) f16 Kt[BKV * LDP];
  __shared__ __align__(16) f16 Vt[CH * LDP];
  __shared__ __align__(16) f16 Ps[NWAVE * 16 * LDP];

  const int tid  = threadIdx.x;
  const int wave = tid >> 5;       // 0..7
  const int lane = tid & 31;
  const int n    = lane & 15;      // fragment column within 16
  const int h16  = lane >> 4;      // lane-half selects K sub-range / row+8

  const int head = blockIdx.y;               // 0..63  (= b*16 + h)
  const int q0   = blockIdx.x * BQ;          // query tile base
  const float* base = qkv + (size_t)head * (3 * CH) * T_SEQ;

  const float scale = 0.35355339059327373f;  // ch^-0.25, applied to q AND k

  // ---- stage Q tile: Qs[t][c] = q[c][q0+t] * scale, f32->f16 --------------
  #pragma unroll 4
  for (int e = tid; e < CH * BQ; e += 256) {
    int c = e >> 7;                 // 0..63
    int t = e & 127;                // coalesced along t
    float v = base[c * T_SEQ + q0 + t] * scale;
    Qs[t * LDP + c] = (f16)v;
  }
  __syncthreads();

  // ---- per-wave Q A-fragments (M=16 rows, two K=32 chunks over ch=64) -----
  const f16* qrow = Qs + (wave * 16 + n) * LDP;   // lane row m = lane&15
  v16h qA0 = load_a_frag(qrow, 0, h16);
  v16h qA1 = load_a_frag(qrow, 32, h16);

  // ---- online softmax state (rows j and j+8*h16 per C/D layout) -----------
  float mrow[8], lrow[8];
  v8f o[4];
  #pragma unroll
  for (int j = 0; j < 8; ++j) { mrow[j] = -3.0e38f; lrow[j] = 0.0f; }
  #pragma unroll
  for (int ct = 0; ct < 4; ++ct) o[ct] = (v8f){0,0,0,0,0,0,0,0};

  for (int kv = 0; kv < T_SEQ / BKV; ++kv) {
    const int s0 = kv * BKV;

    // prefetch next K/V tile into L2 (global_prefetch_b8)
    if (kv + 1 < T_SEQ / BKV) {
      int c = tid & 63;
      const float* nk = base + (size_t)(CH + c) * T_SEQ + s0 + BKV;
      const float* nv = base + (size_t)(2 * CH + c) * T_SEQ + s0 + BKV;
      __builtin_prefetch(nk, 0, 1);
      __builtin_prefetch(nv, 0, 1);
    }

    __syncthreads();   // previous tile fully consumed before overwrite
    // ---- stage K (transposed) and V tiles -------------------------------
    #pragma unroll 4
    for (int e = tid; e < CH * BKV; e += 256) {
      int c = e >> 6;               // 0..63
      int s = e & 63;               // coalesced along s
      float kvval = base[(size_t)(CH + c) * T_SEQ + s0 + s] * scale;
      Kt[s * LDP + c] = (f16)kvval;                 // Kt[s][c]
      float vvval = base[(size_t)(2 * CH + c) * T_SEQ + s0 + s];
      Vt[c * LDP + s] = (f16)vvval;                 // Vt[c][s]
    }
    __syncthreads();

    // ---- GEMM1: S[16 x 64] = Q(16x64) * K^T, 4 N-tiles x 2 K-chunks ------
    v8f sfrag[4];
    #pragma unroll
    for (int nt = 0; nt < 4; ++nt) {
      const f16* kcol = Kt + (nt * 16 + n) * LDP;   // column s = nt*16+n
      v16h kB0 = load_b_frag(kcol + 0  + h16 * 16);
      v16h kB1 = load_b_frag(kcol + 32 + h16 * 16);
      v8f acc = (v8f){0,0,0,0,0,0,0,0};
      acc = wmma_f16(qA0, kB0, acc);
      acc = wmma_f16(qA1, kB1, acc);
      sfrag[nt] = acc;
    }

    // ---- online softmax over this 64-key block ---------------------------
    float bm[8];
    #pragma unroll
    for (int j = 0; j < 8; ++j)
      bm[j] = fmaxf(fmaxf(sfrag[0][j], sfrag[1][j]),
                    fmaxf(sfrag[2][j], sfrag[3][j]));
    #pragma unroll
    for (int j = 0; j < 8; ++j) {
      #pragma unroll
      for (int off = 1; off < 16; off <<= 1)
        bm[j] = fmaxf(bm[j], __shfl_xor(bm[j], off, 32));
    }

    float mnew[8], alpha[8], rs[8];
    #pragma unroll
    for (int j = 0; j < 8; ++j) {
      mnew[j]  = fmaxf(mrow[j], bm[j]);
      alpha[j] = __expf(mrow[j] - mnew[j]);
      mrow[j]  = mnew[j];
      rs[j]    = 0.0f;
    }
    #pragma unroll
    for (int nt = 0; nt < 4; ++nt) {
      #pragma unroll
      for (int j = 0; j < 8; ++j) {
        float p = __expf(sfrag[nt][j] - mnew[j]);
        sfrag[nt][j] = p;
        rs[j] += p;
      }
    }
    #pragma unroll
    for (int j = 0; j < 8; ++j) {
      #pragma unroll
      for (int off = 1; off < 16; off <<= 1)
        rs[j] += __shfl_xor(rs[j], off, 32);
      lrow[j] = lrow[j] * alpha[j] + rs[j];
    }
    #pragma unroll
    for (int ct = 0; ct < 4; ++ct) {
      #pragma unroll
      for (int j = 0; j < 8; ++j) o[ct][j] *= alpha[j];
    }

    // ---- re-layout P via wave-private LDS (DS ops in-order per wave) -----
    f16* pbase = Ps + wave * 16 * LDP;
    #pragma unroll
    for (int nt = 0; nt < 4; ++nt) {
      #pragma unroll
      for (int j = 0; j < 8; ++j) {
        int row = j + 8 * h16;                      // C/D layout row
        pbase[row * LDP + nt * 16 + n] = (f16)sfrag[nt][j];
      }
    }
    const f16* prow = pbase + n * LDP;              // A-frag row m = lane&15
    v16h pA0 = load_a_frag(prow, 0, h16);
    v16h pA1 = load_a_frag(prow, 32, h16);

    // ---- GEMM2: O[16 x 64ch] += P(16x64) * V^T(64 x 64ch) ----------------
    #pragma unroll
    for (int ct = 0; ct < 4; ++ct) {
      const f16* vcol = Vt + (ct * 16 + n) * LDP;   // column c = ct*16+n
      v16h vB0 = load_b_frag(vcol + 0  + h16 * 16);
      v16h vB1 = load_b_frag(vcol + 32 + h16 * 16);
      o[ct] = wmma_f16(pA0, vB0, o[ct]);
      o[ct] = wmma_f16(pA1, vB1, o[ct]);
    }
  }

  // ---- normalize and write out[(head*64 + c) * 1024 + t] -------------------
  #pragma unroll
  for (int ct = 0; ct < 4; ++ct) {
    #pragma unroll
    for (int j = 0; j < 8; ++j) {
      int t = q0 + wave * 16 + j + 8 * h16;
      int c = ct * 16 + n;
      float inv = 1.0f / lrow[j];
      out[((size_t)head * CH + c) * T_SEQ + t] = o[ct][j] * inv;
    }
  }
}

extern "C" void kernel_launch(void* const* d_in, const int* in_sizes, int n_in,
                              void* d_out, int out_size, void* d_ws, size_t ws_size,
                              hipStream_t stream) {
  (void)in_sizes; (void)n_in; (void)out_size; (void)d_ws; (void)ws_size;
  const float* qkv = (const float*)d_in[0];
  float* out = (float*)d_out;
  dim3 grid(T_SEQ / BQ, 64);   // 8 query tiles x 64 heads
  dim3 block(256);             // 8 wave32s
  qkv_attn_kernel<<<grid, block, 0, stream>>>(qkv, out);
}
